// EstraNet_37288906064394
// MI455X (gfx1250) — matrix operand order repaired and checked by reading, not compiled, not run on hardware
//
#include <hip/hip_runtime.h>

// ---------------------------------------------------------------------------
// EstraNet forward for MI455X (gfx1250). All heavy GEMMs via
// v_wmma_f32_16x16x32_f16 (f16 in, f32 accumulate), 2x2 register-blocked:
// each wave computes a 32x32 tile -> 4 WMMAs per k-step for 8 b128 loads,
// doubling arithmetic intensity vs 1 tile/wave. Working set is L2-resident
// (192MB), so fragments stream from L2; no LDS staging needed.
// ---------------------------------------------------------------------------

typedef __attribute__((ext_vector_type(16))) _Float16 v16h;
typedef __attribute__((ext_vector_type(8)))  _Float16 v8h;
typedef __attribute__((ext_vector_type(8)))  float    v8f;

#define BATCHN 32
#define SIGLEN 100000
#define LSEQ   1000
#define BLROWS (BATCHN*LSEQ)     // 32000
#define DMODEL 256
#define NH     8
#define DHEAD  32
#define MF     128               // random features
#define M2     256               // 2*MF (sin|cos)
#define LP     1024              // padded L for K-dim of kv GEMM
#define KSZ    300
#define KSP    320               // padded stem K
#define DFF    1024
#define KSEEDS 4
#define NHP    4
#define HDP    64

// =============================== WMMA GEMM =================================
// C[bt] = act( A[bt](MxK,f16) * B[bt]^T(NxK,f16) * scaleN + biasN ) (*rowscale)
// (+ resid). Batch offsets: off = (bt/div)*so + (bt%div)*si for A/B/C.
// One wave computes a 32x32 tile (2x2 WMMA register blocking).
__global__ __launch_bounds__(256) void k_gemm(
    const _Float16* __restrict__ A, long long sAo, long long sAi, int divA,
    const _Float16* __restrict__ Bt, long long sBo, long long sBi, int divB,
    float* Cf, _Float16* Ch, long long sCo, long long sCi, int divC, int ldc,
    const float* scaleN, const float* biasN,
    const float* rowscale, int rsDiv, int rsStride,
    const float* resid,
    int Mdim, int Ndim, int K, int act)
{
  const int wave = threadIdx.x >> 5;
  const int lane = threadIdx.x & 31;
  const int bt   = blockIdx.y;
  const int tilesN = (Ndim + 31) >> 5;
  const int tilesM = (Mdim + 31) >> 5;
  const int t = blockIdx.x * 8 + wave;
  if (t >= tilesM * tilesN) return;               // uniform per wave: EXEC all-1s
  const int tm = t / tilesN, tn = t % tilesN;
  const int hi = lane >> 4, r16 = lane & 15;

  const _Float16* Ab = A  + (long long)(bt / divA) * sAo + (long long)(bt % divA) * sAi;
  const _Float16* Bb = Bt + (long long)(bt / divB) * sBo + (long long)(bt % divB) * sBi;
  const long long coff = (long long)(bt / divC) * sCo + (long long)(bt % divC) * sCi;

  int arow0 = tm * 32 + r16;      if (arow0 >= Mdim) arow0 = Mdim - 1; // clamp tail
  int arow1 = tm * 32 + 16 + r16; if (arow1 >= Mdim) arow1 = Mdim - 1;
  int bcol0 = tn * 32 + r16;      if (bcol0 >= Ndim) bcol0 = Ndim - 1;
  int bcol1 = tn * 32 + 16 + r16; if (bcol1 >= Ndim) bcol1 = Ndim - 1;
  // A k order used here (not ISA table order): lane holds 16 contiguous K at base hi*16 (matches B)
  const _Float16* Ap0 = Ab + (long long)arow0 * K + hi * 16;
  const _Float16* Ap1 = Ab + (long long)arow1 * K + hi * 16;
  // B k order used here (same as A): lane n holds 16 contiguous K at base hi*16 (Bt is NxK)
  const _Float16* Bp0 = Bb + (long long)bcol0 * K + hi * 16;
  const _Float16* Bp1 = Bb + (long long)bcol1 * K + hi * 16;

  v8f acc00 = {}, acc01 = {}, acc10 = {}, acc11 = {};
  for (int k0 = 0; k0 < K; k0 += 32) {
    if (k0 + 32 < K) {                                  // gfx1250 global_prefetch_b8
      __builtin_prefetch(Ap0 + k0 + 32, 0, 1);
      __builtin_prefetch(Ap1 + k0 + 32, 0, 1);
      __builtin_prefetch(Bp0 + k0 + 32, 0, 1);
      __builtin_prefetch(Bp1 + k0 + 32, 0, 1);
    }
    v8h a0lo = *(const v8h*)(Ap0 + k0);
    v8h a0hi = *(const v8h*)(Ap0 + k0 + 8);
    v8h a1lo = *(const v8h*)(Ap1 + k0);
    v8h a1hi = *(const v8h*)(Ap1 + k0 + 8);
    v16h a0, a1;
#pragma unroll
    for (int i = 0; i < 8; ++i) {
      a0[i] = a0lo[i]; a0[i + 8] = a0hi[i];
      a1[i] = a1lo[i]; a1[i + 8] = a1hi[i];
    }
    v16h b0 = *(const v16h*)(Bp0 + k0);
    v16h b1 = *(const v16h*)(Bp1 + k0);
    acc00 = __builtin_amdgcn_wmma_f32_16x16x32_f16(false, a0, false, b0, (short)0, acc00, false, false);
    acc01 = __builtin_amdgcn_wmma_f32_16x16x32_f16(false, a0, false, b1, (short)0, acc01, false, false);
    acc10 = __builtin_amdgcn_wmma_f32_16x16x32_f16(false, a1, false, b0, (short)0, acc10, false, false);
    acc11 = __builtin_amdgcn_wmma_f32_16x16x32_f16(false, a1, false, b1, (short)0, acc11, false, false);
  }

  // C layout per 16x16 tile: VGPR r -> row base+hi*8+r, col = base + (lane&15)
#pragma unroll
  for (int sn = 0; sn < 2; ++sn) {
    int col = tn * 32 + sn * 16 + r16;
    if (col >= Ndim) continue;
    float sc = scaleN ? scaleN[col] : 1.0f;
    float bi = biasN  ? biasN[col]  : 0.0f;
#pragma unroll
    for (int sm = 0; sm < 2; ++sm) {
      const v8f& acc = (sm == 0) ? (sn == 0 ? acc00 : acc01)
                                 : (sn == 0 ? acc10 : acc11);
#pragma unroll
      for (int r = 0; r < 8; ++r) {
        int row = tm * 32 + sm * 16 + hi * 8 + r;
        if (row < Mdim) {
          float v = acc[r] * sc + bi;
          if (rowscale) v *= rowscale[(long long)(bt % rsDiv) * rsStride + row];
          if (act == 1)      v = fmaxf(v, 0.0f);
          else if (act == 2) v = 0.5f * v * (1.0f + erff(v * 0.70710678118654752f));
          long long off = coff + (long long)row * ldc + col;
          if (resid) v += resid[off];
          if (Cf) Cf[off] = v;
          if (Ch) Ch[off] = (_Float16)v;
        }
      }
    }
  }
}

// ============================ helper kernels ===============================
__global__ void k_cvt(const float* in, _Float16* out, int N, int K, int Kp) {
  long long i = (long long)blockIdx.x * 256 + threadIdx.x;
  if (i >= (long long)N * Kp) return;
  int n = (int)(i / Kp), k = (int)(i % Kp);
  out[i] = (k < K) ? (_Float16)in[(long long)n * K + k] : (_Float16)0.0f;
}

__global__ void k_im2col(const float* x, _Float16* out) {
  long long i = (long long)blockIdx.x * 256 + threadIdx.x;
  if (i >= (long long)BLROWS * KSP) return;
  int kp = (int)(i % KSP);
  long long bl = i / KSP;
  int b = (int)(bl / LSEQ), p = (int)(bl % LSEQ);
  int idx = p * 100 - 100 + kp;             // stride 100, pad 100
  float v = 0.0f;
  if (kp < KSZ && idx >= 0 && idx < SIGLEN) v = x[(long long)b * SIGLEN + idx];
  out[i] = (_Float16)v;
}

__global__ void k_stem_prep(const float* stem_b, const float* g, const float* bb,
                            const float* rm, const float* rv, float* sc, float* bi) {
  int c = blockIdx.x * 256 + threadIdx.x;
  if (c >= DMODEL) return;
  float s = g[c] * rsqrtf(rv[c] + 1e-5f);
  sc[c] = s;
  bi[c] = (stem_b[c] - rm[c]) * s + bb[c];
}

__global__ __launch_bounds__(256) void k_ln(const float* x, const float* g,
                                            const float* b, float* outf,
                                            _Float16* outh, int rows) {
  int wave = threadIdx.x >> 5, lane = threadIdx.x & 31;
  int row = blockIdx.x * 8 + wave;
  if (row >= rows) return;
  const float* xr = x + (long long)row * DMODEL;
  float v[8]; float s = 0.f;
#pragma unroll
  for (int i = 0; i < 8; ++i) { v[i] = xr[lane + i * 32]; s += v[i]; }
#pragma unroll
  for (int m = 16; m >= 1; m >>= 1) s += __shfl_xor(s, m, 32);
  float mu = s * (1.0f / DMODEL);
  float q = 0.f;
#pragma unroll
  for (int i = 0; i < 8; ++i) { float d = v[i] - mu; q += d * d; }
#pragma unroll
  for (int m = 16; m >= 1; m >>= 1) q += __shfl_xor(q, m, 32);
  float inv = rsqrtf(q * (1.0f / DMODEL) + 1e-5f);
#pragma unroll
  for (int i = 0; i < 8; ++i) {
    int c = lane + i * 32;
    float y = (v[i] - mu) * inv * g[c] + b[c];
    long long off = (long long)row * DMODEL + c;
    if (outf) outf[off] = y;
    if (outh) outh[off] = (_Float16)y;
  }
}

// pproj/sproj = pos_ft/pos_slopes @ Wp   (positional features generated inline)
__global__ void k_posproj(const float* Wp, float* pproj, float* sproj) {
  int i = blockIdx.x * 256 + threadIdx.x;
  if (i >= LSEQ * DMODEL) return;
  int o = i % DMODEL, l = i / DMODEL;
  float accp = 0.f, accs = 0.f;
  for (int j = 0; j < DMODEL; ++j) {
    int g = j >> 6, id = j & 63;
    float ns  = (256.0f - 4.0f * id) * (100.0f / 256.0f) / 999.0f;
    float fwd = l * ns, bwd = (999 - l) * ns;
    float pf = (g == 0) ? fwd : (g == 1) ? bwd : (g == 2) ? -fwd : -bwd;
    float ps = ((g == 0 || g == 3) ? ns : -ns) * 999.0f;
    float w = Wp[j * DMODEL + o];
    accp += pf * w; accs += ps * w;
  }
  pproj[i] = accp; sproj[i] = accs;
}

__global__ __launch_bounds__(256) void k_posqk(const float* pproj, const float* sproj,
                                               const float* scale, const float* offp,
                                               float* qpos, float* kpos, float* nf) {
  int wave = threadIdx.x >> 5, d = threadIdx.x & 31;
  int idx = blockIdx.x * 8 + wave;
  if (idx >= LSEQ * NH) return;
  int l = idx >> 3, h = idx & 7;
  float p  = pproj[l * DMODEL + h * DHEAD + d];
  float sp = sproj[l * DMODEL + h * DHEAD + d];
  float sc = scale[h], of = offp[h];
  float qp = sc * p;
  float kp = qp + of * (sc * sp);
  qpos[(l * NH + h) * DHEAD + d] = qp;
  kpos[(l * NH + h) * DHEAD + d] = kp;
  float r = sp * sp;
#pragma unroll
  for (int m = 16; m >= 1; m >>= 1) r += __shfl_xor(r, m, 32);
  if (d == 0) nf[h * LSEQ + l] = sqrtf(r) / (float)LSEQ;
}

// favor+ features: qpH (H,L,M2) row-major f16; kpT (H,M2,LP) f16 (zero padded)
__global__ void k_fkernel(const float* qpos, const float* kpos, const float* proj,
                          _Float16* qpH, _Float16* kpT) {
  int i = blockIdx.x * 256 + threadIdx.x;
  if (i >= NH * MF * LP) return;          // 2^20
  int l = i & (LP - 1);
  int m = (i >> 10) & (MF - 1);
  int h = i >> 17;
  _Float16* krow0 = kpT + ((long long)(h * M2 + m)) * LP + l;
  _Float16* krow1 = kpT + ((long long)(h * M2 + m + MF)) * LP + l;
  if (l >= LSEQ) { *krow0 = (_Float16)0.f; *krow1 = (_Float16)0.f; return; }
  const float dn = 0.42044820762685725f;     // 32^-0.25
  const float ratio = 0.08838834764831845f;  // 128^-0.5
  const float* qv = qpos + (l * NH + h) * DHEAD;
  const float* kv = kpos + (l * NH + h) * DHEAD;
  const float* pr = proj + m * DHEAD;
  float dq = 0.f, dk = 0.f;
#pragma unroll
  for (int d = 0; d < DHEAD; ++d) { dq += qv[d] * pr[d]; dk += kv[d] * pr[d]; }
  dq *= dn; dk *= dn;
  qpH[((long long)(h * LSEQ + l)) * M2 + m]      = (_Float16)(ratio * sinf(dq));
  qpH[((long long)(h * LSEQ + l)) * M2 + m + MF] = (_Float16)(ratio * cosf(dq));
  *krow0 = (_Float16)(ratio * sinf(dk));
  *krow1 = (_Float16)(ratio * cosf(dk));
}

// v (b,l,h,d) f16 -> vT (b,h,d,LP) f16, zero-padded l
__global__ void k_trv(const _Float16* vf, _Float16* vT) {
  int i = blockIdx.x * 256 + threadIdx.x;   // 2^23
  if (i >= BATCHN * NH * DHEAD * LP) return;
  int lp = i & (LP - 1);
  int d  = (i >> 10) & (DHEAD - 1);
  int h  = (i >> 15) & (NH - 1);
  int b  = i >> 18;
  _Float16 v = (_Float16)0.f;
  if (lp < LSEQ) v = vf[(((long long)(b * LSEQ + lp)) * NH + h) * DHEAD + d];
  vT[i] = v;
}

// kv (bt,m,d) f32 -> kvT (bt,d,m) f16
__global__ void k_trkv(const float* kv, _Float16* kvT) {
  int i = blockIdx.x * 256 + threadIdx.x;   // 2^21
  if (i >= 256 * DHEAD * M2) return;
  int m  = i & 255;
  int d  = (i >> 8) & 31;
  int bt = i >> 13;
  kvT[i] = (_Float16)kv[((long long)(bt * M2 + m)) * DHEAD + d];
}

// PMA score: sc[b,h,kq,l]
__global__ void k_scores(const float* q, const float* k, float* sc) {
  int i = blockIdx.x * 256 + threadIdx.x;
  if (i >= BATCHN * NHP * KSEEDS * LSEQ) return;
  int l  = i % LSEQ;
  int kq = (i / LSEQ) % KSEEDS;
  int h  = (i / (LSEQ * KSEEDS)) % NHP;
  int b  = i / (LSEQ * KSEEDS * NHP);
  const float* qp = q + kq * DMODEL + h * HDP;
  const float* kp = k + ((long long)(b * LSEQ + l)) * DMODEL + h * HDP;
  float s = 0.f;
  for (int e = 0; e < HDP; ++e) s += qp[e] * kp[e];
  sc[i] = s * 0.125f;                      // 1/sqrt(64)
}

__global__ __launch_bounds__(256) void k_softmax(float* sc, int n) {
  int row = blockIdx.x;
  float* x = sc + (long long)row * n;
  __shared__ float red[256];
  int tid = threadIdx.x;
  float m = -3.4e38f;
  for (int i = tid; i < n; i += 256) m = fmaxf(m, x[i]);
  red[tid] = m; __syncthreads();
  for (int s = 128; s >= 1; s >>= 1) { if (tid < s) red[tid] = fmaxf(red[tid], red[tid + s]); __syncthreads(); }
  m = red[0]; __syncthreads();
  float sum = 0.f;
  for (int i = tid; i < n; i += 256) { float e = expf(x[i] - m); x[i] = e; sum += e; }
  red[tid] = sum; __syncthreads();
  for (int s = 128; s >= 1; s >>= 1) { if (tid < s) red[tid] += red[tid + s]; __syncthreads(); }
  float inv = 1.0f / red[0];
  for (int i = tid; i < n; i += 256) x[i] *= inv;
}

__global__ void k_attn_o(const float* a, const float* vv, float* o) {
  int i = blockIdx.x * 256 + threadIdx.x;
  if (i >= BATCHN * KSEEDS * NHP * HDP) return;
  int e  = i % HDP;
  int h  = (i / HDP) % NHP;
  int kq = (i / (HDP * NHP)) % KSEEDS;
  int b  = i / (HDP * NHP * KSEEDS);
  const float* ar = a + (((long long)(b * NHP + h)) * KSEEDS + kq) * LSEQ;
  const float* vr = vv + (long long)b * LSEQ * DMODEL + h * HDP + e;
  float s = 0.f;
  for (int l = 0; l < LSEQ; ++l) s += ar[l] * vr[(long long)l * DMODEL];
  o[((long long)(b * KSEEDS + kq)) * DMODEL + h * HDP + e] = s;
}

// small f32 GEMM: out = act(A(MxK)@W(NxK)^T + bias) [+ resid[(m%resMod)*N+n]]
__global__ void k_matvec(const float* A, int lda, const float* W, int ldw,
                         const float* bias, const float* resid, int resMod,
                         float* out, int M, int N, int K, int act) {
  int i = blockIdx.x * 256 + threadIdx.x;
  if (i >= M * N) return;
  int n = i % N, m = i / N;
  const float* ar = A + (long long)m * lda;
  const float* wr = W + (long long)n * ldw;
  float s = bias ? bias[n] : 0.f;
  for (int k = 0; k < K; ++k) s += ar[k] * wr[k];
  if (act == 1) s = fmaxf(s, 0.f);
  if (resid) s += resid[(long long)(m % resMod) * N + n];
  out[i] = s;
}

__global__ __launch_bounds__(256) void k_logsoftmax(const float* in, float* out, int n) {
  int row = blockIdx.x;
  const float* x = in + (long long)row * n;
  __shared__ float red[256];
  int tid = threadIdx.x;
  float m = -3.4e38f;
  for (int i = tid; i < n; i += 256) m = fmaxf(m, x[i]);
  red[tid] = m; __syncthreads();
  for (int s = 128; s >= 1; s >>= 1) { if (tid < s) red[tid] = fmaxf(red[tid], red[tid + s]); __syncthreads(); }
  m = red[0]; __syncthreads();
  float sum = 0.f;
  for (int i = tid; i < n; i += 256) sum += expf(x[i] - m);
  red[tid] = sum; __syncthreads();
  for (int s = 128; s >= 1; s >>= 1) { if (tid < s) red[tid] += red[tid + s]; __syncthreads(); }
  float lse = m + logf(red[0]);
  for (int i = tid; i < n; i += 256) out[(long long)row * n + i] = x[i] - lse;
}

// ============================ host orchestration ===========================
static inline void gemm(hipStream_t st,
                        const _Float16* A, long long sAo, long long sAi, int divA,
                        const _Float16* Bt, long long sBo, long long sBi, int divB,
                        float* Cf, _Float16* Ch, long long sCo, long long sCi,
                        int divC, int ldc,
                        const float* scaleN, const float* biasN,
                        const float* rs, int rsDiv, int rsStride,
                        const float* resid, int M, int N, int K, int act, int nbatch) {
  int tiles = ((M + 31) / 32) * ((N + 31) / 32);
  dim3 g((unsigned)((tiles + 7) / 8), (unsigned)nbatch, 1);
  k_gemm<<<g, 256, 0, st>>>(A, sAo, sAi, divA, Bt, sBo, sBi, divB,
                            Cf, Ch, sCo, sCi, divC, ldc, scaleN, biasN,
                            rs, rsDiv, rsStride, resid, M, N, K, act);
}

#define GRID1(n) dim3((unsigned)(((long long)(n) + 255) / 256))

// workspace offsets (bytes). total required ~200MB.
static constexpr size_t O_H     = 0;                       // h f32      32.77MB
static constexpr size_t O_HH    = 33000000;                // h f16      16.38MB
static constexpr size_t O_TMP   = 50000000;                // tmp f32    32.77MB
static constexpr size_t O_BIG   = 83000000;                // multi-use  65.54MB
static constexpr size_t O_Z16   = O_BIG + 33554432;        // z f16 (during layers)
static constexpr size_t O_VBUF  = O_BIG + 32768000;        // PMA vv f32 (after layers)
static constexpr size_t O_PPROJ = 149000000;
static constexpr size_t O_SPROJ = 150100000;
static constexpr size_t O_QPOS  = 151200000;
static constexpr size_t O_KPOS  = 152300000;
static constexpr size_t O_NF    = 153400000;
static constexpr size_t O_QPH   = 153500000;               // (H,L,256) f16
static constexpr size_t O_KPT   = 157700000;               // (H,256,LP) f16
static constexpr size_t O_VT    = 162000000;               // (B,H,32,LP) f16
static constexpr size_t O_KV    = 179000000;               // (256,256,32) f32
static constexpr size_t O_KVT   = 187500000;               // (256,32,256) f16
static constexpr size_t O_WC    = 191800000;               // converted f16 weights
static constexpr size_t O_SB    = 195900000;               // stem scale/bias
static constexpr size_t O_PMA   = 196000000;               // small PMA/cls buffers

extern "C" void kernel_launch(void* const* d_in, const int* in_sizes, int n_in,
                              void* d_out, int out_size, void* d_ws, size_t ws_size,
                              hipStream_t stream) {
  (void)in_sizes; (void)n_in; (void)out_size; (void)ws_size;
  const float* x      = (const float*)d_in[0];
  const float* stem_w = (const float*)d_in[1];
  const float* stem_b = (const float*)d_in[2];
  const float* bn_g   = (const float*)d_in[3];
  const float* bn_b   = (const float*)d_in[4];
  const float* bn_rm  = (const float*)d_in[5];
  const float* bn_rv  = (const float*)d_in[6];
  // layer params: 7 + layer*14 + {0:Wv 1:Wo 2:Wp 3:scale 4:off 5:proj
  //                               6:ln1_g 7:ln1_b 8:ln2_g 9:ln2_b
  //                               10:f1_w 11:f1_b 12:f2_w 13:f2_b}
  const int base2 = 7 + 2 * 14;   // 35
  const float* seeds     = (const float*)d_in[base2 + 0];
  const float* mha_in_w  = (const float*)d_in[base2 + 1];
  const float* mha_in_b  = (const float*)d_in[base2 + 2];
  const float* mha_out_w = (const float*)d_in[base2 + 3];
  const float* mha_out_b = (const float*)d_in[base2 + 4];
  const float* pma_ln1_g = (const float*)d_in[base2 + 5];
  const float* pma_ln1_b = (const float*)d_in[base2 + 6];
  const float* pma_ln2_g = (const float*)d_in[base2 + 7];
  const float* pma_ln2_b = (const float*)d_in[base2 + 8];
  const float* pma_f1_w  = (const float*)d_in[base2 + 9];
  const float* pma_f1_b  = (const float*)d_in[base2 + 10];
  const float* pma_f2_w  = (const float*)d_in[base2 + 11];
  const float* pma_f2_b  = (const float*)d_in[base2 + 12];
  const float* cls_w1    = (const float*)d_in[base2 + 13];
  const float* cls_b1    = (const float*)d_in[base2 + 14];
  const float* cls_w2    = (const float*)d_in[base2 + 15];
  const float* cls_b2    = (const float*)d_in[base2 + 16];

  char* ws = (char*)d_ws;
  float*     Hf    = (float*)(ws + O_H);
  _Float16*  HH    = (_Float16*)(ws + O_HH);
  float*     TMP   = (float*)(ws + O_TMP);
  _Float16*  PATCH = (_Float16*)(ws + O_BIG);     // im2col (stem only)
  _Float16*  VF16  = (_Float16*)(ws + O_BIG);     // v f16 (layers)
  _Float16*  FF16  = (_Float16*)(ws + O_BIG);     // ffn hidden f16 (layers)
  float*     KBUF  = (float*)(ws + O_BIG);        // PMA k f32 (tail)
  _Float16*  ZF16  = (_Float16*)(ws + O_Z16);
  float*     VBUF  = (float*)(ws + O_VBUF);       // PMA vv f32 (tail)
  float*     PPROJ = (float*)(ws + O_PPROJ);
  float*     SPROJ = (float*)(ws + O_SPROJ);
  float*     QPOS  = (float*)(ws + O_QPOS);
  float*     KPOS  = (float*)(ws + O_KPOS);
  float*     NF    = (float*)(ws + O_NF);
  _Float16*  QPH   = (_Float16*)(ws + O_QPH);
  _Float16*  KPT   = (_Float16*)(ws + O_KPT);
  _Float16*  VT    = (_Float16*)(ws + O_VT);
  float*     KV    = (float*)(ws + O_KV);
  _Float16*  KVT   = (_Float16*)(ws + O_KVT);
  _Float16*  WSTEM = (_Float16*)(ws + O_WC);
  float*     SBSC  = (float*)(ws + O_SB);
  float*     SBBI  = (float*)(ws + O_SB + 1024);
  _Float16*  WKH   = (_Float16*)(ws + O_WC + 3100000);
  _Float16*  WV2H  = (_Float16*)(ws + O_WC + 3300000);
  float* QBUF = (float*)(ws + O_PMA);
  float* SCB  = (float*)(ws + O_PMA + 16384);
  float* OBUF = (float*)(ws + O_PMA + 2100000);
  float* OUT1 = (float*)(ws + O_PMA + 2250000);
  float* OUT2 = (float*)(ws + O_PMA + 2400000);
  float* FFB  = (float*)(ws + O_PMA + 2550000);
  float* OUT3 = (float*)(ws + O_PMA + 3100000);
  float* PMAO = (float*)(ws + O_PMA + 3250000);
  float* HCB  = (float*)(ws + O_PMA + 3400000);
  float* LOGB = (float*)(ws + O_PMA + 3450000);

  // ---- stem: im2col GEMM + folded BN + exact GELU ----
  k_im2col<<<GRID1((long long)BLROWS * KSP), 256, 0, stream>>>(x, PATCH);
  k_cvt<<<GRID1(DMODEL * KSP), 256, 0, stream>>>(stem_w, WSTEM, DMODEL, KSZ, KSP);
  k_stem_prep<<<1, 256, 0, stream>>>(stem_b, bn_g, bn_b, bn_rm, bn_rv, SBSC, SBBI);
  gemm(stream, PATCH, 0, 0, 1, WSTEM, 0, 0, 1, Hf, HH, 0, 0, 1, DMODEL,
       SBSC, SBBI, nullptr, 1, 0, nullptr, BLROWS, DMODEL, KSP, /*gelu*/2, 1);

  // ---- 2 linear-attention transformer layers ----
  for (int l = 0; l < 2; ++l) {
    const float* Wv   = (const float*)d_in[7 + l * 14 + 0];
    const float* Wo   = (const float*)d_in[7 + l * 14 + 1];
    const float* Wp   = (const float*)d_in[7 + l * 14 + 2];
    const float* scal = (const float*)d_in[7 + l * 14 + 3];
    const float* offp = (const float*)d_in[7 + l * 14 + 4];
    const float* proj = (const float*)d_in[7 + l * 14 + 5];
    const float* ln1g = (const float*)d_in[7 + l * 14 + 6];
    const float* ln1b = (const float*)d_in[7 + l * 14 + 7];
    const float* ln2g = (const float*)d_in[7 + l * 14 + 8];
    const float* ln2b = (const float*)d_in[7 + l * 14 + 9];
    const float* f1w  = (const float*)d_in[7 + l * 14 + 10];
    const float* f1b  = (const float*)d_in[7 + l * 14 + 11];
    const float* f2w  = (const float*)d_in[7 + l * 14 + 12];
    const float* f2b  = (const float*)d_in[7 + l * 14 + 13];

    size_t wb = O_WC + 200000 + (size_t)l * 1400000;
    _Float16* WvH = (_Float16*)(ws + wb);
    _Float16* WoH = (_Float16*)(ws + wb + 140000);
    _Float16* F1H = (_Float16*)(ws + wb + 280000);
    _Float16* F2H = (_Float16*)(ws + wb + 804352);
    k_cvt<<<GRID1(DMODEL * DMODEL), 256, 0, stream>>>(Wv, WvH, DMODEL, DMODEL, DMODEL);
    k_cvt<<<GRID1(DMODEL * DMODEL), 256, 0, stream>>>(Wo, WoH, DMODEL, DMODEL, DMODEL);
    k_cvt<<<GRID1(DFF * DMODEL), 256, 0, stream>>>(f1w, F1H, DFF, DMODEL, DMODEL);
    k_cvt<<<GRID1(DMODEL * DFF), 256, 0, stream>>>(f2w, F2H, DMODEL, DFF, DFF);

    // positional features -> qpos/kpos/nf -> favor+ feature maps
    k_posproj<<<GRID1(LSEQ * DMODEL), 256, 0, stream>>>(Wp, PPROJ, SPROJ);
    k_posqk<<<GRID1(LSEQ * NH * 32), 256, 0, stream>>>(PPROJ, SPROJ, scal, offp,
                                                       QPOS, KPOS, NF);
    k_fkernel<<<GRID1(NH * MF * LP), 256, 0, stream>>>(QPOS, KPOS, proj, QPH, KPT);

    // v = h @ Wv^T   (f16 out)
    gemm(stream, HH, 0, 0, 1, WvH, 0, 0, 1, nullptr, VF16, 0, 0, 1, DMODEL,
         nullptr, nullptr, nullptr, 1, 0, nullptr, BLROWS, DMODEL, DMODEL, 0, 1);
    k_trv<<<GRID1(BATCHN * NH * DHEAD * LP), 256, 0, stream>>>(VF16, VT);

    // kv[b,h] = k'^T @ v : batched (256x32, K=LP) over bt=b*H+h
    gemm(stream, KPT, 0, (long long)M2 * LP, NH,
         VT, (long long)DHEAD * LP, 0, 1,
         KV, nullptr, (long long)M2 * DHEAD, 0, 1, DHEAD,
         nullptr, nullptr, nullptr, 1, 0, nullptr, M2, DHEAD, LP, 0, BATCHN * NH);
    k_trkv<<<GRID1(256 * DHEAD * M2), 256, 0, stream>>>(KV, KVT);

    // z[b,h] = q' @ kv, scaled by nf -> f16 (b,l,h*32+d)
    gemm(stream, QPH, 0, (long long)LSEQ * M2, NH,
         KVT, (long long)DHEAD * M2, 0, 1,
         nullptr, ZF16, (long long)LSEQ * DMODEL, DHEAD, NH, DMODEL,
         nullptr, nullptr, NF, NH, LSEQ, nullptr,
         LSEQ, DHEAD, M2, 0, BATCHN * NH);

    // attn_out = z @ Wo^T + h ; LN1
    gemm(stream, ZF16, 0, 0, 1, WoH, 0, 0, 1, TMP, nullptr, 0, 0, 1, DMODEL,
         nullptr, nullptr, nullptr, 1, 0, Hf, BLROWS, DMODEL, DMODEL, 0, 1);
    k_ln<<<dim3(BLROWS / 8), 256, 0, stream>>>(TMP, ln1g, ln1b, Hf, HH, BLROWS);

    // FFN: relu(h@f1^T+b1)@f2^T+b2 + h ; LN2
    gemm(stream, HH, 0, 0, 1, F1H, 0, 0, 1, nullptr, FF16, 0, 0, 1, DFF,
         nullptr, f1b, nullptr, 1, 0, nullptr, BLROWS, DFF, DMODEL, /*relu*/1, 1);
    gemm(stream, FF16, 0, 0, 1, F2H, 0, 0, 1, TMP, nullptr, 0, 0, 1, DMODEL,
         nullptr, f2b, nullptr, 1, 0, Hf, BLROWS, DMODEL, DFF, 0, 1);
    k_ln<<<dim3(BLROWS / 8), 256, 0, stream>>>(TMP, ln2g, ln2b, Hf, HH, BLROWS);
  }

  // ---- PMA (4 seed queries) + classifier tail ----
  k_cvt<<<GRID1(DMODEL * DMODEL), 256, 0, stream>>>(mha_in_w + 256 * 256, WKH,
                                                    DMODEL, DMODEL, DMODEL);
  k_cvt<<<GRID1(DMODEL * DMODEL), 256, 0, stream>>>(mha_in_w + 512 * 256, WV2H,
                                                    DMODEL, DMODEL, DMODEL);
  gemm(stream, HH, 0, 0, 1, WKH, 0, 0, 1, KBUF, nullptr, 0, 0, 1, DMODEL,
       nullptr, mha_in_b + 256, nullptr, 1, 0, nullptr, BLROWS, DMODEL, DMODEL, 0, 1);
  gemm(stream, HH, 0, 0, 1, WV2H, 0, 0, 1, VBUF, nullptr, 0, 0, 1, DMODEL,
       nullptr, mha_in_b + 512, nullptr, 1, 0, nullptr, BLROWS, DMODEL, DMODEL, 0, 1);
  k_matvec<<<GRID1(KSEEDS * DMODEL), 256, 0, stream>>>(
      seeds, DMODEL, mha_in_w, DMODEL, mha_in_b, nullptr, 1, QBUF,
      KSEEDS, DMODEL, DMODEL, 0);
  k_scores<<<GRID1(BATCHN * NHP * KSEEDS * LSEQ), 256, 0, stream>>>(QBUF, KBUF, SCB);
  k_softmax<<<dim3(BATCHN * NHP * KSEEDS), 256, 0, stream>>>(SCB, LSEQ);
  k_attn_o<<<GRID1(BATCHN * KSEEDS * NHP * HDP), 256, 0, stream>>>(SCB, VBUF, OBUF);
  k_matvec<<<GRID1(128 * DMODEL), 256, 0, stream>>>(
      OBUF, DMODEL, mha_out_w, DMODEL, mha_out_b, seeds, KSEEDS, OUT1,
      128, DMODEL, DMODEL, 0);
  k_ln<<<dim3(16), 256, 0, stream>>>(OUT1, pma_ln1_g, pma_ln1_b, OUT2, nullptr, 128);
  k_matvec<<<GRID1(128 * DFF), 256, 0, stream>>>(
      OUT2, DMODEL, pma_f1_w, DMODEL, pma_f1_b, nullptr, 1, FFB, 128, DFF, DMODEL, 1);
  k_matvec<<<GRID1(128 * DMODEL), 256, 0, stream>>>(
      FFB, DFF, pma_f2_w, DFF, pma_f2_b, OUT2, 128, OUT3, 128, DMODEL, DFF, 0);
  k_ln<<<dim3(16), 256, 0, stream>>>(OUT3, pma_ln2_g, pma_ln2_b, PMAO, nullptr, 128);
  // PMAO contiguous (128,256) == (32,1024)
  k_matvec<<<GRID1(32 * DMODEL), 256, 0, stream>>>(
      PMAO, 1024, cls_w1, 1024, cls_b1, nullptr, 1, HCB, 32, DMODEL, 1024, 1);
  k_matvec<<<GRID1(32 * DMODEL), 256, 0, stream>>>(
      HCB, DMODEL, cls_w2, DMODEL, cls_b2, nullptr, 1, LOGB, 32, DMODEL, DMODEL, 0);
  k_logsoftmax<<<dim3(32), 256, 0, stream>>>(LOGB, (float*)d_out, DMODEL);
}
